// TALSQLinear_67843303408246
// MI455X (gfx1250) — 8-bit matrix operand order repaired and checked by reading, compiled, not run on hardware
//
#include <hip/hip_runtime.h>
#include <stdint.h>

typedef __attribute__((ext_vector_type(16))) int   v16i;
typedef __attribute__((ext_vector_type(8)))  float v8f;
typedef int v4i_t __attribute__((vector_size(16)));

#define AS1 __attribute__((address_space(1)))
#define AS3 __attribute__((address_space(3)))

#define IN_F   4096
#define OUT_F  4096
#define M_TOT  8192
#define KBYTES 128
#define KT     (IN_F / KBYTES)
#define BM     128
#define BN     64
#define A_STRIDE 144
#define B_STRIDE 144
#define A_BYTES (BM * A_STRIDE)
#define B_BYTES (BN * B_STRIDE)
#define BUF_BYTES (A_BYTES + B_BYTES)

__device__ __forceinline__ void async_copy_b128(const uint8_t* g, uint8_t* l) {
#if __has_builtin(__builtin_amdgcn_global_load_async_to_lds_b128)
    __builtin_amdgcn_global_load_async_to_lds_b128(
        (AS1 v4i_t*)g, (AS3 v4i_t*)l, 0, 0);
#else
    *(uint4*)l = *(const uint4*)g;
#endif
}
__device__ __forceinline__ void wait_async0() {
#if __has_builtin(__builtin_amdgcn_s_wait_asynccnt)
    __builtin_amdgcn_s_wait_asynccnt(0);
#elif __has_builtin(__builtin_amdgcn_global_load_async_to_lds_b128)
    asm volatile("s_wait_asynccnt 0" ::: "memory");
#endif
}

__device__ __forceinline__ uint32_t fp8_q(float v) {
    float q = rintf(v);
    q = fminf(fmaxf(q, -8.0f), 7.0f);
    uint32_t s = 0u;
    if (q < 0.0f) { s = 0x80u; q = -q; }
    int mi = (int)q;
    uint32_t e;
    if (mi < 4)      e = (0x44403800u >> (8 * mi)) & 0xFFu;
    else if (mi < 8) e = (0x4E4C4A48u >> (8 * (mi - 4))) & 0xFFu;
    else             e = 0x50u;
    return s | e;
}

__global__ __launch_bounds__(256)
void talsq_quant_w(const float* __restrict__ W,  const float* __restrict__ lA,
                   const float* __restrict__ lB, const float* __restrict__ wscale,
                   uint32_t* __restrict__ qw)
{
    const int o   = blockIdx.x;
    const int tid = threadIdx.x;
    float b[8];
#pragma unroll
    for (int r = 0; r < 8; ++r) b[r] = lB[o * 8 + r];
    const float sc = wscale[o];
    const float* Wrow = W + (size_t)o * IN_F;
    uint32_t*    qrow = qw + (size_t)o * (IN_F / 4);
#pragma unroll
    for (int c = 0; c < 4; ++c) {
        const int i4 = tid + 256 * c;
        const int i  = i4 * 4;
        float4 w = *(const float4*)(Wrow + i);
        float l0 = 0.f, l1 = 0.f, l2 = 0.f, l3 = 0.f;
#pragma unroll
        for (int r = 0; r < 8; ++r) {
            float4 a = *(const float4*)(lA + (size_t)r * IN_F + i);
            l0 += b[r] * a.x; l1 += b[r] * a.y; l2 += b[r] * a.z; l3 += b[r] * a.w;
        }
        qrow[i4] =  fp8_q((w.x + l0) / sc)
                 | (fp8_q((w.y + l1) / sc) << 8)
                 | (fp8_q((w.z + l2) / sc) << 16)
                 | (fp8_q((w.w + l3) / sc) << 24);
    }
}

__global__ __launch_bounds__(256)
void talsq_quant_x(const float* __restrict__ x, const float* __restrict__ ts,
                   const int* __restrict__ stepp, uint32_t* __restrict__ qx)
{
    const float sc = ts[*stepp];
    const size_t i4 = (size_t)blockIdx.x * 256 + threadIdx.x;
    const float4 v = *(const float4*)(x + i4 * 4);
    qx[i4] =  fp8_q(v.x / sc)
           | (fp8_q(v.y / sc) << 8)
           | (fp8_q(v.z / sc) << 16)
           | (fp8_q(v.w / sc) << 24);
}

__global__ __launch_bounds__(256, 1)
void talsq_gemm_fp8(const uint8_t* __restrict__ qx, const uint8_t* __restrict__ qw,
                    const float* __restrict__ wscale, const float* __restrict__ tscale,
                    const int* __restrict__ stepp, const float* __restrict__ bias,
                    float* __restrict__ out)
{
    __shared__ __align__(16) uint8_t sm[2 * BUF_BYTES];

    const int tid   = threadIdx.x;
    const int lane  = tid & 31;
    const int wave  = tid >> 5;
    const int waveM = wave & 3;
    const int waveN = wave >> 2;
    const int h     = lane >> 4;
    const int r16   = lane & 15;

    const int nBase = blockIdx.x * BN;
    const int mBase = blockIdx.y * BM;

    const int arow = tid >> 3;
    const int acol = (tid & 7) * 16;
    const uint8_t* gA = qx + (size_t)(mBase + arow) * IN_F + acol;
    const uint8_t* gB = qw + (size_t)(nBase + arow) * IN_F + acol;
    const uint32_t ldsAoff = (uint32_t)arow * A_STRIDE + acol;
    const uint32_t ldsBoff = (uint32_t)arow * B_STRIDE + acol;

    auto issueTile = [&](int kOff, int buf) {
        uint8_t* bufA = sm + buf * BUF_BYTES;
        uint8_t* bufB = bufA + A_BYTES;
#pragma unroll
        for (int j = 0; j < 4; ++j)
            async_copy_b128(gA + (size_t)j * 32 * IN_F + kOff,
                            bufA + ldsAoff + j * 32 * A_STRIDE);
#pragma unroll
        for (int j = 0; j < 2; ++j)
            async_copy_b128(gB + (size_t)j * 32 * IN_F + kOff,
                            bufB + ldsBoff + j * 32 * B_STRIDE);
    };

    v8f acc[2][2];
#pragma unroll
    for (int t = 0; t < 2; ++t)
#pragma unroll
        for (int s = 0; s < 2; ++s) acc[t][s] = (v8f){0,0,0,0,0,0,0,0};

    issueTile(0, 0);
    wait_async0();
    __syncthreads();

    for (int kb = 0; kb < KT; ++kb) {
        const int cur = kb & 1;
        const uint8_t* bufA = sm + cur * BUF_BYTES;
        const uint8_t* bufB = bufA + A_BYTES;

        if (kb + 1 < KT) issueTile((kb + 1) * KBYTES, cur ^ 1);

        v16i af[2], bf[2];
#pragma unroll
        for (int t = 0; t < 2; ++t) {
            const uint8_t* p = bufA + (waveM * 32 + t * 16 + r16) * A_STRIDE + 16 * h;
#pragma unroll
            for (int j = 0; j < 8; ++j) {
                uint2 d = *(const uint2*)(p + 32 * (j >> 1) + 8 * (j & 1));
                af[t][2 * j]     = (int)d.x;
                af[t][2 * j + 1] = (int)d.y;
            }
        }
#pragma unroll
        for (int s = 0; s < 2; ++s) {
            const uint8_t* p = bufB + (waveN * 32 + s * 16 + r16) * B_STRIDE + 16 * h;
#pragma unroll
            for (int c = 0; c < 4; ++c) {
                uint4 d = *(const uint4*)(p + 32 * c);
                bf[s][4 * c]     = (int)d.x;
                bf[s][4 * c + 1] = (int)d.y;
                bf[s][4 * c + 2] = (int)d.z;
                bf[s][4 * c + 3] = (int)d.w;
            }
        }

#pragma unroll
        for (int t = 0; t < 2; ++t)
#pragma unroll
            for (int s = 0; s < 2; ++s)
                acc[t][s] = __builtin_amdgcn_wmma_f32_16x16x128_fp8_fp8(
                    af[t], bf[s], (short)0, acc[t][s], false, false);

        wait_async0();
        __syncthreads();
    }

    const float xs = tscale[*stepp];
#pragma unroll
    for (int t = 0; t < 2; ++t) {
#pragma unroll
        for (int s = 0; s < 2; ++s) {
            const int n  = nBase + waveN * 32 + s * 16 + r16;
            const float scl = wscale[n] * xs;
            const float bv  = bias[n];
            const int m0 = mBase + waveM * 32 + t * 16 + 8 * h;
            float* po = out + (size_t)m0 * OUT_F + n;
#pragma unroll
            for (int r = 0; r < 8; ++r)
                po[(size_t)r * OUT_F] = acc[t][s][r] * scl + bv;
        }
    }
}

extern "C" void kernel_launch(void* const* d_in, const int* in_sizes, int n_in,
                              void* d_out, int out_size, void* d_ws, size_t ws_size,
                              hipStream_t stream) {
    (void)in_sizes; (void)n_in; (void)out_size; (void)ws_size;
    const float* x    = (const float*)d_in[0];
    const float* W    = (const float*)d_in[1];
    const float* lA   = (const float*)d_in[2];
    const float* lB   = (const float*)d_in[3];
    const float* ws   = (const float*)d_in[4];
    const float* ts   = (const float*)d_in[5];
    const float* bias = (const float*)d_in[6];
    const int*   step = (const int*)d_in[7];
    float* out = (float*)d_out;

    uint8_t* qw = (uint8_t*)d_ws;
    uint8_t* qx = qw + (size_t)OUT_F * IN_F;

    talsq_quant_w<<<OUT_F, 256, 0, stream>>>(W, lA, lB, ws, (uint32_t*)qw);

    const int xBlocks = (int)(((size_t)M_TOT * IN_F / 4) / 256);
    talsq_quant_x<<<xBlocks, 256, 0, stream>>>(x, ts, step, (uint32_t*)qx);

    talsq_gemm_fp8<<<dim3(OUT_F / BN, M_TOT / BM), 256, 0, stream>>>(
        qx, qw, ws, ts, step, bias, out);
}
